// NMPSFHiFi_70669391889225
// MI455X (gfx1250) — matrix operand order repaired and checked by reading, not compiled, not run on hardware
//
#include <hip/hip_runtime.h>
#include <hip/hip_bf16.h>
#include <math.h>

// ---------------- problem constants (match reference) ----------------
constexpr int   B_    = 4;
constexpr int   NF_   = 512;
constexpr int   NU_   = 256;
constexpr int   NH_   = 256;
constexpr int   BLK_  = 512;
constexpr int   WIN_  = 2048;
constexpr int   NBIN_ = WIN_/2 + 1;          // 1025
constexpr int   T_    = NF_*BLK_;            // 262144
constexpr int   NFR_  = T_/BLK_ + 1;         // 513 stft frames
constexpr int   TOTY_ = (NFR_-1)*BLK_ + WIN_;// 264192
constexpr int   TOT_  = 2*NBIN_ + 256;       // 2306 ctrl cols
constexpr int   K1P_  = 288;                 // 260 padded to /32
constexpr int   NROW_ = B_*NF_;              // 2048 MLP rows
constexpr int   N2P_  = 2320;                // 2306 padded to /16
constexpr int   NOISE_LEN_ = WIN_*127;       // 260096
constexpr float SR_   = 44100.0f;
constexpr float PI_   = 3.14159265358979323846f;
constexpr float F0V_  = 0.1f;

typedef __attribute__((ext_vector_type(16))) _Float16 v16h;
typedef __attribute__((ext_vector_type(8)))  float    v8f;
union F16x16 { v16h v; float4 f[2]; };

// ---------------- utility ----------------
__global__ void k_zero(float* __restrict__ p, long n) {
  long i = (long)blockIdx.x * blockDim.x + threadIdx.x;
  long stride = (long)gridDim.x * blockDim.x;
  for (; i < n; i += stride) p[i] = 0.0f;
}

// ---------------- 1) f0 per-sample + fractional cumulative phase ----------------
__global__ void k_scan(const float* __restrict__ f0f, const float* __restrict__ fvar,
                       float* __restrict__ x) {
  const int b   = blockIdx.x;
  const int tid = threadIdx.x;
  const int lane = tid & 31, wid = tid >> 5;
  __shared__ float warpsum[8];
  __shared__ float s_carry;
  if (tid == 0) s_carry = 0.0f;
  __syncthreads();
  const float base_det = exp2f(-F0V_ / 12.0f);
  for (int base = 0; base < T_; base += 256) {
    int t = base + tid;
    float f0 = f0f[b*NF_ + (t >> 9)];                 // BLK_=512
    float v  = f0 * base_det * exp2f(fvar[(long)b*T_ + t] * (F0V_/12.0f)) / SR_;
    // wave32 inclusive scan
    #pragma unroll
    for (int d = 1; d < 32; d <<= 1) {
      float u = __shfl_up(v, d, 32);
      if (lane >= d) v += u;
    }
    if (lane == 31) warpsum[wid] = v;
    __syncthreads();
    if (tid == 0) {                                   // exclusive scan of wave totals
      float a = 0.0f;
      #pragma unroll
      for (int w = 0; w < 8; ++w) { float tt = warpsum[w]; warpsum[w] = a; a += tt; }
    }
    __syncthreads();
    float cs = s_carry + warpsum[wid] + v;
    x[(long)b*T_ + t] = cs - floorf(cs);
    __syncthreads();
    if (tid == 255) s_carry = cs - floorf(cs);        // keep only frac for precision
    __syncthreads();
  }
}

// ---------------- 2) comb impulse train + per-frame phase ----------------
__global__ void k_comb_phase(const float* __restrict__ x, float* __restrict__ comb,
                             float* __restrict__ sinb, float* __restrict__ cosb) {
  long i = (long)blockIdx.x * blockDim.x + threadIdx.x;
  if (i >= (long)B_*T_) return;
  int b = (int)(i / T_), t = (int)(i % T_);
  float x0 = x[i];
  float x1 = x[(long)b*T_ + ((t + 1) == T_ ? 0 : t + 1)];   // jnp.roll wraps
  comb[i] = (x1 - x0 < 0.0f) ? 1.0f : 0.0f;
  if ((t & (BLK_-1)) == 0) {
    int fr = t >> 9;
    float ph = 2.0f * PI_ * x0;
    sinb[b*NF_ + fr] = sinf(ph);
    cosb[b*NF_ + fr] = cosf(ph);
  }
}

// ---------------- 3) MLP input assembly (f16, K padded) ----------------
__global__ void k_build_A1(const float* __restrict__ units, const float* __restrict__ f0f,
                           const float* __restrict__ vol, const float* __restrict__ sinb,
                           const float* __restrict__ cosb, _Float16* __restrict__ A1) {
  long i = (long)blockIdx.x * blockDim.x + threadIdx.x;
  if (i >= (long)NROW_ * K1P_) return;
  int row = (int)(i / K1P_), col = (int)(i % K1P_);
  float v = 0.0f;
  if      (col < NU_)  v = units[(long)row*NU_ + col];
  else if (col == 256) v = f0f[row] / 700.0f;
  else if (col == 257) v = vol[row];
  else if (col == 258) v = sinb[row];
  else if (col == 259) v = cosb[row];
  A1[i] = (_Float16)v;
}

__global__ void k_w1t(const float* __restrict__ W1, _Float16* __restrict__ W1t) {
  long i = (long)blockIdx.x * blockDim.x + threadIdx.x;
  if (i >= (long)NH_ * K1P_) return;
  int n = (int)(i / K1P_), k = (int)(i % K1P_);
  W1t[i] = (k < NU_ + 4) ? (_Float16)W1[(long)k*NH_ + n] : (_Float16)0.0f;
}

__global__ void k_w2t(const float* __restrict__ W2, _Float16* __restrict__ W2t) {
  long i = (long)blockIdx.x * blockDim.x + threadIdx.x;
  if (i >= (long)N2P_ * NH_) return;
  int n = (int)(i / NH_), k = (int)(i % NH_);
  W2t[i] = (n < TOT_) ? (_Float16)W2[(long)k*TOT_ + n] : (_Float16)0.0f;
}

// ---------------- 4) WMMA GEMM: C = A(row-major) * Bt(N-major)^T + bias ----------------
// One 16x16 tile per wave32; K in steps of 32 via v_wmma_f32_16x16x32_f16.
template<int KDIM, int NDIM, bool TANH_F16OUT>
__global__ void k_gemm(const _Float16* __restrict__ A, const _Float16* __restrict__ Bt,
                       const float* __restrict__ bias, void* __restrict__ out, int biasN) {
  const int lane = threadIdx.x & 31;
  const int wave = threadIdx.x >> 5;
  const int tile = blockIdx.x * (blockDim.x >> 5) + wave;
  const int tiles_n = NDIM / 16;
  const int tm = tile / tiles_n, tn = tile % tiles_n;
  if (tm * 16 >= NROW_) return;
  const int r  = lane & 15;
  const int hi = lane >> 4;
  // A 16x32 f16 fragment: lanes<16 hold K k..k+7 / k+16..k+23 of row r; lanes>=16 hold +8
  const _Float16* arow = A  + (long)(tm*16 + r)*KDIM + hi*8;
  // B 32x16 f16 fragment from Bt (row n = output col, contiguous K): same K layout as A (lanes>=16 hold +8)
  const _Float16* brow = Bt + (long)(tn*16 + r)*KDIM + hi*8;
  v8f acc = {};
  #pragma unroll 4
  for (int k = 0; k < KDIM; k += 32) {
    F16x16 af, bf;
    af.f[0] = *(const float4*)(arow + k);
    af.f[1] = *(const float4*)(arow + k + 16);
    bf.f[0] = *(const float4*)(brow + k);
    bf.f[1] = *(const float4*)(brow + k + 16);
    acc = __builtin_amdgcn_wmma_f32_16x16x32_f16(false, af.v, false, bf.v,
                                                 (short)0, acc, false, false);
  }
  const int n  = tn*16 + r;
  const float bv = (n < biasN) ? bias[n] : 0.0f;
  #pragma unroll
  for (int i = 0; i < 8; ++i) {
    int m = tm*16 + i + hi*8;
    float v = acc[i] + bv;
    if constexpr (TANH_F16OUT) ((_Float16*)out)[(long)m*NDIM + n] = (_Float16)tanhf(v);
    else                       ((float*)out)[(long)m*NDIM + n]   = v;
  }
}

// ---------------- 5) window^2 overlap sum (static, no atomics) ----------------
__global__ void k_wsq(const float* __restrict__ window, float* __restrict__ wsq) {
  int t = blockIdx.x * blockDim.x + threadIdx.x;
  if (t >= TOTY_) return;
  float a = 0.0f;
  int fhi = t / BLK_;
  int flo = fhi - 3; if (flo < 0) flo = 0;
  if (fhi > NFR_-1) fhi = NFR_-1;
  for (int f = flo; f <= fhi; ++f) {
    int j = t - f*BLK_;
    if (j >= 0 && j < WIN_) { float w = window[j]; a += w*w; }
  }
  wsq[t] = a;
}

// ---------------- 6) 2048-pt radix-2 FFT in LDS ----------------
__device__ inline void fft_lds(float* re, float* im, float sign) {
  const int tid = threadIdx.x;
  __syncthreads();
  for (int i = tid; i < WIN_; i += 256) {
    unsigned j = __brev((unsigned)i) >> (32 - 11);
    if (j > (unsigned)i) {
      float tr = re[i]; re[i] = re[j]; re[j] = tr;
      float ti = im[i]; im[i] = im[j]; im[j] = ti;
    }
  }
  __syncthreads();
  #pragma unroll 1
  for (int s = 1; s <= 11; ++s) {
    const int mh = 1 << (s-1);
    for (int k = tid; k < WIN_/2; k += 256) {
      int grp = k >> (s-1), pos = k & (mh-1);
      int i0 = (grp << s) + pos, i1 = i0 + mh;
      float ang = sign * (2.0f * PI_) * (float)pos / (float)(mh << 1);
      float sn, c; __sincosf(ang, &sn, &c);
      float br = re[i1], bi = im[i1];
      float xr = br*c - bi*sn;
      float xi = br*sn + bi*c;
      re[i1] = re[i0] - xr; im[i1] = im[i0] - xi;
      re[i0] += xr;         im[i0] += xi;
    }
    __syncthreads();
  }
}

// ---------------- 7) STFT -> filter -> ISTFT (overlap-add), one frame/block ----------
// mode 0: comb * fmw * src_filter (per b,frame).  mode 1: tiled static noise * fmw_env.
__global__ void k_stft(const float* __restrict__ sig, const float* __restrict__ noise,
                       const float* __restrict__ window, const float* __restrict__ fmw,
                       const float* __restrict__ fmw_env, const float* __restrict__ ctrl,
                       float* __restrict__ yout, int mode) {
  __shared__ float sre[WIN_];
  __shared__ float sim[WIN_];
  const int tid = threadIdx.x;
  const int b = (mode == 0) ? (blockIdx.x / NFR_) : 0;
  const int f = (mode == 0) ? (blockIdx.x % NFR_) : blockIdx.x;
  // windowed frame (center pad = WIN/2)
  for (int i = tid; i < WIN_; i += 256) {
    int t = f*BLK_ + i - WIN_/2;
    float s = 0.0f;
    if (t >= 0 && t < T_)
      s = (mode == 0) ? sig[(long)b*T_ + t] : noise[t % NOISE_LEN_];
    sre[i] = s * window[i];
    sim[i] = 0.0f;
  }
  fft_lds(sre, sim, -1.0f);
  // apply filter on bins 0..1024
  const float* crow = nullptr;
  if (mode == 0) {
    int fr = (f < NF_) ? f : NF_-1;                   // src_filter last-frame repeat
    crow = ctrl + (long)(b*NF_ + fr)*N2P_;
  }
  const float* F = (mode == 0) ? fmw : fmw_env;       // complex64: interleaved re,im
  for (int k = tid; k < NBIN_; k += 256) {
    float Fr = F[2*k], Fi = F[2*k+1];
    if (mode == 0) {
      float e = expf(crow[k]);
      float sp, cp; __sincosf(PI_ * crow[NBIN_ + k], &sp, &cp);
      float sr2 = e*cp, si2 = e*sp;                   // exp(mag + i*pi*ph)
      float tr = Fr*sr2 - Fi*si2, ti = Fr*si2 + Fi*sr2;
      Fr = tr; Fi = ti;
    }
    float xr = sre[k], xi = sim[k];
    sre[k] = xr*Fr - xi*Fi;
    sim[k] = xr*Fi + xi*Fr;
  }
  __syncthreads();
  // hermitian mirror -> full spectrum
  for (int k = 1 + tid; k < WIN_/2; k += 256) {
    sre[WIN_-k] =  sre[k];
    sim[WIN_-k] = -sim[k];
  }
  fft_lds(sre, sim, +1.0f);
  const float inv = 1.0f / (float)WIN_;
  float* yb = yout + (mode == 0 ? (long)b*TOTY_ : 0);
  for (int i = tid; i < WIN_; i += 256)
    atomicAdd(&yb[f*BLK_ + i], sre[i] * inv * window[i]);
}

// ---------------- 8) envelope upsample + source mix ----------------
__global__ void k_source(const float* __restrict__ ycomb, const float* __restrict__ ynoise,
                         const float* __restrict__ wsq, const float* __restrict__ ctrl,
                         float* __restrict__ src) {
  long i = (long)blockIdx.x * blockDim.x + threadIdx.x;
  if (i >= (long)B_*T_) return;
  int b = (int)(i / T_), t = (int)(i % T_);
  float wq = fmaxf(wsq[WIN_/2 + t], 1e-8f);
  float c  = ycomb[(long)b*TOTY_ + WIN_/2 + t] / wq;
  float n  = ynoise[WIN_/2 + t] / wq;
  const int ENV0 = 2*NBIN_ + 128;                     // 2178 = nenv start
  int fi  = t >> 2;
  int fi1 = fi + 1; if (fi1 > NF_*128 - 1) fi1 = NF_*128 - 1;
  float e0 = expf(ctrl[(long)(b*NF_ + (fi  >> 7))*N2P_ + ENV0 + (fi  & 127)]);
  float e1 = expf(ctrl[(long)(b*NF_ + (fi1 >> 7))*N2P_ + ENV0 + (fi1 & 127)]);
  float frac = (float)(t & 3) * 0.25f;
  src[i] = c + n * (e0 * (1.0f - frac) + e1 * frac);
}

__device__ inline float lrelu(float v) { return v > 0.0f ? v : 0.1f * v; }

// ---------------- 9) conv stack stage 0: nsf(128,NF) x8 -> 64ch @4096 ----------------
__global__ void k_conv0(const float* __restrict__ ctrl, const float* __restrict__ Wg0,
                        float* __restrict__ g1) {
  long i = (long)blockIdx.x * blockDim.x + threadIdx.x;
  if (i >= (long)B_*64*4096) return;
  int t  = (int)(i & 4095);
  int co = (int)((i >> 12) & 63);
  int b  = (int)(i >> 18);
  const int NSF0 = 2*NBIN_;                           // 2050
  float acc = 0.0f;
  #pragma unroll
  for (int dk = 0; dk < 3; ++dk) {
    int u = t + dk - 1;
    if (u < 0 || u >= 4096) continue;
    const float* cr = ctrl + (long)(b*NF_ + (u >> 3))*N2P_ + NSF0;
    const float* w  = Wg0 + (long)co*128*3 + dk;
    float a = 0.0f;
    for (int ci = 0; ci < 128; ++ci) a += w[ci*3] * cr[ci];
    acc += a;
  }
  g1[i] = lrelu(acc);
}

// ---------------- 10) conv stage 1: 64ch x8 -> 32ch @32768 ----------------
__global__ void k_conv1(const float* __restrict__ g1, const float* __restrict__ Wg1,
                        float* __restrict__ g2) {
  long i = (long)blockIdx.x * blockDim.x + threadIdx.x;
  if (i >= (long)B_*32*32768) return;
  int t  = (int)(i & 32767);
  int co = (int)((i >> 15) & 31);
  int b  = (int)(i >> 20);
  float acc = 0.0f;
  #pragma unroll
  for (int dk = 0; dk < 3; ++dk) {
    int u = t + dk - 1;
    if (u < 0 || u >= 32768) continue;
    const float* gr = g1 + (long)b*64*4096 + (u >> 3);
    const float* w  = Wg1 + (long)co*64*3 + dk;
    float a = 0.0f;
    for (int ci = 0; ci < 64; ++ci) a += w[ci*3] * gr[(long)ci*4096];
    acc += a;
  }
  g2[i] = lrelu(acc);
}

// ---------------- 11) fused tail: conv2(32->16,x8) + Wsrc + Wout + tanh ----------------
__global__ void k_tail(const float* __restrict__ g2, const float* __restrict__ src,
                       const float* __restrict__ Wg2, const float* __restrict__ Wsrc,
                       const float* __restrict__ Wout, float* __restrict__ out) {
  __shared__ float g[16][260];                        // positions t0-1 .. t0+256
  const int b  = blockIdx.y;
  const int t0 = blockIdx.x * 256;
  const int tid = threadIdx.x;
  for (int it = tid; it < 16*258; it += 256) {
    int co = it / 258, p = it % 258;
    int t = t0 - 1 + p;
    float v = 0.0f;
    if (t >= 0 && t < T_) {
      float acc = 0.0f;
      #pragma unroll
      for (int dk = 0; dk < 3; ++dk) {
        int u = t + dk - 1;
        if (u < 0 || u >= T_) continue;
        const float* gr = g2 + (long)b*32*32768 + (u >> 3);
        const float* w  = Wg2 + (long)co*32*3 + dk;
        float a = 0.0f;
        for (int ci = 0; ci < 32; ++ci) a += w[ci*3] * gr[(long)ci*32768];
        acc += a;
      }
      v = lrelu(acc) + Wsrc[co] * src[(long)b*T_ + t];
    }
    g[co][p] = v;
  }
  __syncthreads();
  float o = 0.0f;
  #pragma unroll
  for (int co = 0; co < 16; ++co)
    #pragma unroll
    for (int dk = 0; dk < 3; ++dk)
      o += Wout[co*3 + dk] * g[co][tid + dk];
  out[(long)b*T_ + t0 + tid] = tanhf(o);
}

// ---------------- host-side launch ----------------
extern "C" void kernel_launch(void* const* d_in, const int* in_sizes, int n_in,
                              void* d_out, int out_size, void* d_ws, size_t ws_size,
                              hipStream_t stream) {
  const float* units = (const float*)d_in[0];
  const float* f0f   = (const float*)d_in[1];
  const float* vol   = (const float*)d_in[2];
  const float* fvar  = (const float*)d_in[3];
  const float* win   = (const float*)d_in[4];
  const float* fmw   = (const float*)d_in[5];   // complex64 -> interleaved f32
  const float* fmwE  = (const float*)d_in[6];   // complex64 -> interleaved f32
  const float* nse   = (const float*)d_in[7];
  const float* W1    = (const float*)d_in[8];
  const float* b1    = (const float*)d_in[9];
  const float* W2    = (const float*)d_in[10];
  const float* b2    = (const float*)d_in[11];
  const float* Wg0   = (const float*)d_in[12];
  const float* Wg1   = (const float*)d_in[13];
  const float* Wg2   = (const float*)d_in[14];
  const float* Wsrc  = (const float*)d_in[15];
  const float* Wout  = (const float*)d_in[16];
  float* out = (float*)d_out;

  size_t off = 0;
  auto alloc = [&](size_t bytes) -> void* {
    void* p = (char*)d_ws + off;
    off += (bytes + 255) & ~(size_t)255;
    return p;
  };
  float*     x      = (float*)    alloc(sizeof(float)    * (size_t)B_*T_);
  float*     comb   = (float*)    alloc(sizeof(float)    * (size_t)B_*T_);
  float*     sinb   = (float*)    alloc(sizeof(float)    * (size_t)B_*NF_);
  float*     cosb   = (float*)    alloc(sizeof(float)    * (size_t)B_*NF_);
  _Float16*  A1     = (_Float16*) alloc(sizeof(_Float16) * (size_t)NROW_*K1P_);
  _Float16*  W1t    = (_Float16*) alloc(sizeof(_Float16) * (size_t)NH_*K1P_);
  _Float16*  Hb     = (_Float16*) alloc(sizeof(_Float16) * (size_t)NROW_*NH_);
  _Float16*  W2t    = (_Float16*) alloc(sizeof(_Float16) * (size_t)N2P_*NH_);
  float*     ctrl   = (float*)    alloc(sizeof(float)    * (size_t)NROW_*N2P_);
  float*     ycomb  = (float*)    alloc(sizeof(float)    * (size_t)B_*TOTY_);
  float*     ynoise = (float*)    alloc(sizeof(float)    * (size_t)TOTY_);
  float*     wsq    = (float*)    alloc(sizeof(float)    * (size_t)TOTY_);
  float*     src    = (float*)    alloc(sizeof(float)    * (size_t)B_*T_);
  float*     g1     = (float*)    alloc(sizeof(float)    * (size_t)B_*64*4096);
  float*     g2     = (float*)    alloc(sizeof(float)    * (size_t)B_*32*32768);
  (void)ws_size; (void)n_in; (void)in_sizes; (void)out_size;

  auto cdiv = [](long a, long b) -> int { return (int)((a + b - 1) / b); };

  // zero the overlap-add accumulators (atomics target)
  k_zero<<<1024, 256, 0, stream>>>(ycomb, (long)B_*TOTY_);
  k_zero<<<1024, 256, 0, stream>>>(ynoise, (long)TOTY_);

  // phase scan, comb train, frame phase
  k_scan<<<B_, 256, 0, stream>>>(f0f, fvar, x);
  k_comb_phase<<<cdiv((long)B_*T_, 256), 256, 0, stream>>>(x, comb, sinb, cosb);

  // MLP: build f16 operands, two WMMA GEMMs
  k_build_A1<<<cdiv((long)NROW_*K1P_, 256), 256, 0, stream>>>(units, f0f, vol, sinb, cosb, A1);
  k_w1t<<<cdiv((long)NH_*K1P_, 256), 256, 0, stream>>>(W1, W1t);
  k_w2t<<<cdiv((long)N2P_*NH_, 256), 256, 0, stream>>>(W2, W2t);
  // GEMM1: 2048x288x256, tanh -> f16 H    (2048 tiles / 8 waves)
  k_gemm<K1P_, NH_,  true ><<<(NROW_/16)*(NH_/16)/8,  256, 0, stream>>>(A1, W1t, b1, Hb,  NH_);
  // GEMM2: 2048x256x2320 (+bias) -> f32 ctrl (18560 tiles / 8 waves)
  k_gemm<NH_,  N2P_, false><<<(NROW_/16)*(N2P_/16)/8, 256, 0, stream>>>(Hb, W2t, b2, ctrl, TOT_);

  // ISTFT normalization denominator (static)
  k_wsq<<<cdiv(TOTY_, 256), 256, 0, stream>>>(win, wsq);

  // comb branch: per-frame source filter; noise branch: static filter, shared across batch
  k_stft<<<B_*NFR_, 256, 0, stream>>>(comb, nse, win, fmw, fmwE, ctrl, ycomb, 0);
  k_stft<<<NFR_,    256, 0, stream>>>(comb, nse, win, fmw, fmwE, ctrl, ynoise, 1);

  // envelope upsample + mix
  k_source<<<cdiv((long)B_*T_, 256), 256, 0, stream>>>(ycomb, ynoise, wsq, ctrl, src);

  // conv stack
  k_conv0<<<cdiv((long)B_*64*4096, 256), 256, 0, stream>>>(ctrl, Wg0, g1);
  k_conv1<<<cdiv((long)B_*32*32768, 256), 256, 0, stream>>>(g1, Wg1, g2);
  dim3 tg(T_/256, B_);
  k_tail<<<tg, 256, 0, stream>>>(g2, src, Wg2, Wsrc, Wout, out);
}
